// NeuralSpectralBlock2D_24043226923203
// MI455X (gfx1250) — matrix operand order repaired and checked by reading, not compiled, not run on hardware
//
#include <hip/hip_runtime.h>
#include <hip/hip_bf16.h>

typedef _Float16 f16;
typedef __attribute__((ext_vector_type(16))) _Float16 v16h;
typedef __attribute__((ext_vector_type(8)))  _Float16 v8h;
typedef __attribute__((ext_vector_type(8)))  float    v8f;

constexpr int kB     = 16;
constexpr int kC     = 64;
constexpr int kH     = 240;
constexpr int kW     = 240;
constexpr int kHp    = 80;
constexpr int kWp    = 80;
constexpr int kP     = 3;
constexpr int kL     = 9;       // patch area
constexpr int kHEAD  = 8;
constexpr int kDH    = 8;
constexpr int kT     = 4;
constexpr int kM     = 12;

constexpr int kNP    = 16;          // patches per workgroup
constexpr int kNCOL  = kNP * kL;    // 144 GEMM columns (9 col-tiles of 16)
constexpr int kNT    = 256;         // threads (8 wave32)

// LDS column strides (padded, 16B aligned)
constexpr int XPS = 72;   // f16 per column for xp (64 + 8 pad)
constexpr int XQS = 72;   // f16 per column for xq
constexpr int KVS = 136;  // f16 per column for kv (128 + 8 pad)
constexpr int XOS = 68;   // f32 per column for xo

constexpr int OFF_XQ = kNCOL * XPS * 2;              // 20736
constexpr int OFF_LT = OFF_XQ + kNCOL * XQS * 2;     // 41472
constexpr int OFF_KV = OFF_LT + kNP * kC * kT * 4;   // 57856 (lt: 16KB)
constexpr int SMEM_BYTES = OFF_KV + kNCOL * KVS * 2; // 97024 (kv 39168 == xo 39168)

constexpr float kPI = 3.14159265358979323846f;

// 16 f16 from LDS with only 16B alignment guaranteed -> two b128 reads
__device__ __forceinline__ v16h ld_b16x16(const f16* p) {
  union { v16h v; v8h h[2]; } u;
  u.h[0] = *(const v8h*)(p);
  u.h[1] = *(const v8h*)(p + 16);
  return u.v;
}

// A-fragment per CDNA5 16-bit A layout (§7.12.2): lane = (m, hi);
// elem e<8 -> K = 8*hi + e ; e>=8 -> K = 16 + 8*hi + (e-8). kbase in {0,32}.
__device__ __forceinline__ v16h load_a_frag(const float* wmat, int row, int kbase, int hi) {
  v16h a;
#pragma unroll
  for (int e = 0; e < 16; ++e) {
    int k = (e < 8) ? (8 * hi + e) : (16 + 8 * hi + (e - 8));
    a[e] = (f16)wmat[row * kC + kbase + k];
  }
  return a;
}

__global__ __launch_bounds__(kNT, 2)
void nsb2d_fused_kernel(const float* __restrict__ x,
                        const float* __restrict__ latent,
                        const float* __restrict__ weights,
                        const float* __restrict__ enc_w,
                        const float* __restrict__ enc_b,
                        const float* __restrict__ dec_w,
                        const float* __restrict__ dec_b,
                        float* __restrict__ out) {
  extern __shared__ char smem[];
  f16*   xp_h = (f16*)smem;                 // [col][c] f16, stride XPS
  f16*   xq_h = (f16*)(smem + OFF_XQ);      // [col][c] f16, stride XQS
  float* lt_s = (float*)(smem + OFF_LT);    // [p][c][t] f32
  f16*   kv_h = (f16*)(smem + OFF_KV);      // [col][o] f16, stride KVS
  float* xo_s = (float*)(smem + OFF_KV);    // aliases kv (kv dead by then)

  const int tid = threadIdx.x;
  const int g   = blockIdx.x;
  const int gw  = g % (kWp / kNP);
  const int hp  = (g / (kWp / kNP)) % kHp;
  const int bi  = g / ((kWp / kNP) * kHp);
  const int wp0 = gw * kNP;
  const size_t xbase = (size_t)bi * kC * kH * kW;

  // ---------------- Phase 1: stage xp into LDS as f16 (coalesced) ----------
  for (int idx = tid; idx < kC * kNCOL; idx += kNT) {
    int c   = idx / kNCOL;
    int rem = idx - c * kNCOL;
    int i   = rem / (kNP * kP);          // row within patch 0..2
    int r48 = rem - i * (kNP * kP);
    int p   = r48 / kP;                  // patch 0..15
    int jj  = r48 - p * kP;              // col within patch 0..2
    int col = p * kL + i * kP + jj;
    float v = x[xbase + (size_t)c * (kH * kW) + (hp * kP + i) * kW + (wp0 + p) * kP + jj];
    xp_h[col * XPS + c] = (f16)v;
  }
  __syncthreads();

  // ---------------- Phase 2: KV = enc_w(128x64) @ xp(64x144) + enc_b (WMMA) -
  const int wv   = tid >> 5;
  const int lane = tid & 31;
  const int mrow = lane & 15;
  const int hi   = lane >> 4;
  {
    const int obase = wv * 16;                         // wave owns 16 output rows
    v16h a0 = load_a_frag(enc_w, obase + mrow, 0, hi);
    v16h a1 = load_a_frag(enc_w, obase + mrow, 32, hi);
    for (int nt = 0; nt < 9; ++nt) {
      int col = nt * 16 + mrow;                        // B lane = column
      v16h b0 = ld_b16x16(xp_h + col * XPS + hi * 8);        // K 0..31
      v16h b1 = ld_b16x16(xp_h + col * XPS + 32 + hi * 8);   // K 32..63
      v8f acc = {};
      acc = __builtin_amdgcn_wmma_f32_16x16x32_f16(false, a0, false, b0, (short)0, acc, false, false);
      acc = __builtin_amdgcn_wmma_f32_16x16x32_f16(false, a1, false, b1, (short)0, acc, false, false);
#pragma unroll
      for (int r = 0; r < 8; ++r) {                    // D: M = r + 8*hi, N = lane&15
        int oo = obase + hi * 8 + r;
        kv_h[col * KVS + oo] = (f16)(acc[r] + enc_b[oo]);
      }
    }
  }
  __syncthreads();

  // ---------------- Phase 3: latent encoder attention (per patch,head) -----
  if (tid < kNP * kHEAD) {
    const int p = tid >> 3, h = tid & 7;
    float q[kT][kDH];
#pragma unroll
    for (int t4 = 0; t4 < kT; ++t4)
#pragma unroll
      for (int d = 0; d < kDH; ++d)
        q[t4][d] = latent[(h * kT + t4) * kDH + d];
#pragma unroll
    for (int t4 = 0; t4 < kT; ++t4) {
      float sc[kL];
      float mx = -1e30f;
      for (int l = 0; l < kL; ++l) {
        const f16* kcol = kv_h + (p * kL + l) * KVS;
        float s = 0.f;
#pragma unroll
        for (int d = 0; d < kDH; ++d) s += q[t4][d] * (float)kcol[(h * kDH + d) * 2];
        sc[l] = s;
        mx = fmaxf(mx, s);
      }
      float sum = 0.f;
      for (int l = 0; l < kL; ++l) { float e = __expf(sc[l] - mx); sc[l] = e; sum += e; }
      float inv = 1.0f / sum;
#pragma unroll
      for (int d = 0; d < kDH; ++d) {
        float a = 0.f;
        for (int l = 0; l < kL; ++l)
          a += sc[l] * (float)kv_h[(p * kL + l) * KVS + (h * kDH + d) * 2 + 1];
        lt_s[p * (kC * kT) + (h * kDH + d) * kT + t4] = a * inv + q[t4][d];
      }
    }
  }
  __syncthreads();

  // ---------------- Phase 4: spectral basis mix (per patch,channel) --------
  for (int idx = tid; idx < kNP * kC; idx += kNT) {
    int p = idx >> 6, c = idx & 63;
#pragma unroll
    for (int t4 = 0; t4 < kT; ++t4) {
      float a = lt_s[p * (kC * kT) + c * kT + t4];
      float acc = a;
#pragma unroll
      for (int mm = 0; mm < kM; ++mm) {
        float ang = a * (kPI * (float)mm / (float)kM);
        float sn, cs;
        __sincosf(ang, &sn, &cs);
        acc += sn * weights[c * (2 * kM) + mm] + cs * weights[c * (2 * kM) + kM + mm];
      }
      lt_s[p * (kC * kT) + c * kT + t4] = acc;
    }
  }
  __syncthreads();

  // ---------------- Phase 5: XQ = dec_w(64x64) @ xp(64x144) + dec_b (WMMA) -
  for (int tile = wv; tile < 36; tile += 8) {
    int rt = tile / 9, nt = tile - rt * 9;
    int obase = rt * 16;
    v16h a0 = load_a_frag(dec_w, obase + mrow, 0, hi);
    v16h a1 = load_a_frag(dec_w, obase + mrow, 32, hi);
    int col = nt * 16 + mrow;
    v16h b0 = ld_b16x16(xp_h + col * XPS + hi * 8);
    v16h b1 = ld_b16x16(xp_h + col * XPS + 32 + hi * 8);
    v8f acc = {};
    acc = __builtin_amdgcn_wmma_f32_16x16x32_f16(false, a0, false, b0, (short)0, acc, false, false);
    acc = __builtin_amdgcn_wmma_f32_16x16x32_f16(false, a1, false, b1, (short)0, acc, false, false);
#pragma unroll
    for (int r = 0; r < 8; ++r) {
      int oo = obase + hi * 8 + r;
      xq_h[col * XQS + oo] = (f16)(acc[r] + dec_b[oo]);
    }
  }
  __syncthreads();

  // ---------------- Phase 6: decoder attention + residual ------------------
  if (tid < kNP * kHEAD) {
    const int p = tid >> 3, h = tid & 7;
    float lk[kT][kDH];
#pragma unroll
    for (int t4 = 0; t4 < kT; ++t4)
#pragma unroll
      for (int d = 0; d < kDH; ++d)
        lk[t4][d] = lt_s[p * (kC * kT) + (h * kDH + d) * kT + t4];
    for (int l = 0; l < kL; ++l) {
      int col = p * kL + l;
      float xqv[kDH];
#pragma unroll
      for (int d = 0; d < kDH; ++d) xqv[d] = (float)xq_h[col * XQS + h * kDH + d];
      float sc[kT];
      float mx = -1e30f;
#pragma unroll
      for (int t4 = 0; t4 < kT; ++t4) {
        float s = 0.f;
#pragma unroll
        for (int d = 0; d < kDH; ++d) s += xqv[d] * lk[t4][d];
        sc[t4] = s;
        mx = fmaxf(mx, s);
      }
      float sum = 0.f;
#pragma unroll
      for (int t4 = 0; t4 < kT; ++t4) { float e = __expf(sc[t4] - mx); sc[t4] = e; sum += e; }
      float inv = 1.0f / sum;
#pragma unroll
      for (int d = 0; d < kDH; ++d) {
        float a = 0.f;
#pragma unroll
        for (int t4 = 0; t4 < kT; ++t4) a += sc[t4] * lk[t4][d];
        xo_s[col * XOS + h * kDH + d] = a * inv + (float)xp_h[col * XPS + h * kDH + d];
      }
    }
  }
  __syncthreads();

  // ---------------- Phase 7: coalesced writeout (unpatchify) ---------------
  for (int idx = tid; idx < kC * kNCOL; idx += kNT) {
    int c   = idx / kNCOL;
    int rem = idx - c * kNCOL;
    int i   = rem / (kNP * kP);
    int r48 = rem - i * (kNP * kP);
    int p   = r48 / kP;
    int jj  = r48 - p * kP;
    int col = p * kL + i * kP + jj;
    out[xbase + (size_t)c * (kH * kW) + (hp * kP + i) * kW + (wp0 + p) * kP + jj] =
        xo_s[col * XOS + c];
  }
}

extern "C" void kernel_launch(void* const* d_in, const int* in_sizes, int n_in,
                              void* d_out, int out_size, void* d_ws, size_t ws_size,
                              hipStream_t stream) {
  (void)in_sizes; (void)n_in; (void)d_ws; (void)ws_size; (void)out_size;
  const float* x      = (const float*)d_in[0];
  const float* latent = (const float*)d_in[1];
  const float* wts    = (const float*)d_in[2];
  const float* enc_w  = (const float*)d_in[3];
  const float* enc_b  = (const float*)d_in[4];
  const float* dec_w  = (const float*)d_in[5];
  const float* dec_b  = (const float*)d_in[6];
  float* out = (float*)d_out;

  static bool attr_set = false;  // attribute setting is idempotent & capture-safe
  if (!attr_set) {
    hipFuncSetAttribute((const void*)nsb2d_fused_kernel,
                        hipFuncAttributeMaxDynamicSharedMemorySize, SMEM_BYTES);
    attr_set = true;
  }

  dim3 grid(kB * kHp * (kWp / kNP));  // 6400 workgroups of 16 patches
  nsb2d_fused_kernel<<<grid, kNT, SMEM_BYTES, stream>>>(
      x, latent, wts, enc_w, enc_b, dec_w, dec_b, out);
}
